// PolicyNetwork_74964359184976
// MI455X (gfx1250) — matrix operand order repaired and checked by reading, not compiled, not run on hardware
//
#include <hip/hip_runtime.h>
#include <hip/hip_bf16.h>

typedef __attribute__((ext_vector_type(16))) _Float16 v16h;
typedef __attribute__((ext_vector_type(8)))  _Float16 v8h;
typedef __attribute__((ext_vector_type(8)))  float    v8f;

#define BATCH 256
#define NNODE 1024
#define EGRAPH 4096
#define ESWAP 512
#define OBSD 256
#define NACT 600
#define NEGV (-100000000.0f)

// ---------------------------------------------------------------- utilities
__global__ void fill_f32(float* __restrict__ p, float val, size_t n) {
  size_t gid = (size_t)blockIdx.x * blockDim.x + threadIdx.x;
  if (gid < n) p[gid] = val;
}

// Build f16 TRANSPOSED weights: W2t[n][k] (n<128, k<256 = [w2s;w2n]),
// esW1t[n][k] (n<128, k<512). Transposed so a lane's 16-K B-fragment is one
// contiguous 32B chunk -> global_load_b128 pairs instead of 16x b16 loads.
__global__ void cvt_weights(const float* __restrict__ w2s, const float* __restrict__ w2n,
                            const float* __restrict__ esw1,
                            _Float16* __restrict__ W2t, _Float16* __restrict__ esW1t) {
  int gid = blockIdx.x * 256 + threadIdx.x;
  if (gid < 32768) {                       // 256K x 128N
    int k = gid >> 7, n = gid & 127;
    float v = (k < 128) ? w2s[k * 128 + n] : w2n[(k - 128) * 128 + n];
    W2t[(size_t)n * 256 + k] = (_Float16)v;
  } else if (gid < 32768 + 65536) {        // 512K x 128N
    int g2 = gid - 32768;
    int k = g2 >> 7, n = g2 & 127;
    esW1t[(size_t)n * 512 + k] = (_Float16)esw1[k * 128 + n];
  }
}

// ------------------------------------------------ SAGE aggregation (layer 1)
__global__ void scatter_x(const float* __restrict__ x, const int* __restrict__ ei,
                          float* __restrict__ msum1, float* __restrict__ cnt) {
  int gid = blockIdx.x * 256 + threadIdx.x;
  if (gid >= BATCH * EGRAPH) return;
  int e = gid & (EGRAPH - 1);
  int b = gid >> 12;
  int src = ei[(size_t)b * 2 * EGRAPH + e];
  int dst = ei[(size_t)b * 2 * EGRAPH + EGRAPH + e];
  atomicAdd(&cnt[(size_t)b * NNODE + dst], 1.0f);
  size_t sb = ((size_t)b * NNODE + src) * 5;
  size_t db = ((size_t)b * NNODE + dst) * 5;
  #pragma unroll
  for (int k = 0; k < 5; ++k) atomicAdd(&msum1[db + k], x[sb + k]);
}

// h1 = relu(x @ w1s + mean1 @ w1n + b1), stored f16
__global__ __launch_bounds__(256) void sage1(const float* __restrict__ x,
                                             const float* __restrict__ msum1,
                                             const float* __restrict__ cnt,
                                             const float* __restrict__ w1s,
                                             const float* __restrict__ w1n,
                                             const float* __restrict__ b1,
                                             _Float16* __restrict__ h1) {
  int gid = blockIdx.x * 256 + threadIdx.x;     // BATCH*NNODE*128
  int f = gid & 127;
  size_t node = (size_t)(gid >> 7);
  float c = cnt[node]; c = c > 1.0f ? c : 1.0f;
  float acc = b1[f];
  #pragma unroll
  for (int k = 0; k < 5; ++k) {
    acc += x[node * 5 + k] * w1s[k * 128 + f];
    acc += (msum1[node * 5 + k] / c) * w1n[k * 128 + f];
  }
  h1[gid] = (_Float16)(acc > 0.0f ? acc : 0.0f);
}

// scatter h1[src] into msum2[dst] (8 feats per thread)
__global__ void scatter_h1(const _Float16* __restrict__ h1, const int* __restrict__ ei,
                           float* __restrict__ msum2) {
  int gid = blockIdx.x * 256 + threadIdx.x;     // BATCH*EGRAPH*16
  int c = gid & 15;
  int e = (gid >> 4) & (EGRAPH - 1);
  int b = gid >> 16;
  int src = ei[(size_t)b * 2 * EGRAPH + e];
  int dst = ei[(size_t)b * 2 * EGRAPH + EGRAPH + e];
  size_t sb = ((size_t)b * NNODE + src) * 128 + c * 8;
  size_t db = ((size_t)b * NNODE + dst) * 128 + c * 8;
  #pragma unroll
  for (int k = 0; k < 8; ++k) atomicAdd(&msum2[db + k], (float)h1[sb + k]);
}

// ------------------------------------------------ WMMA fragment helpers
// A fragment: lane (g,m) reads 16 consecutive halves of row m, K = K0+16g..+15.
__device__ __forceinline__ v16h load_a_frag(const _Float16* rowPtr, int K0, int g) {
  v8h lo = *(const v8h*)(rowPtr + K0 + 16 * g);
  v8h hi = *(const v8h*)(rowPtr + K0 + 16 * g + 8);
  return __builtin_shufflevector(lo, hi, 0, 1, 2, 3, 4, 5, 6, 7,
                                 8, 9, 10, 11, 12, 13, 14, 15);
}
// B fragment (same contiguous k order as A): lane (g,m) = column colBase+m, K = K0+16g..+15,
// contiguous in the transposed weight buffer.
__device__ __forceinline__ v16h load_b_frag(const _Float16* Wt, int Ktot,
                                            int col, int K0, int g) {
  return *(const v16h*)(Wt + (size_t)col * Ktot + K0 + 16 * g);
}

// ------------------------------------------------ SAGE layer 2 via WMMA
// h2 = relu([h1 | mean2] @ [w2s;w2n] + b2), K = 256
__global__ __launch_bounds__(256) void gnn_layer2_wmma(
    const _Float16* __restrict__ h1, const float* __restrict__ msum2,
    const float* __restrict__ cnt, const _Float16* __restrict__ W2t,
    const float* __restrict__ b2, _Float16* __restrict__ h2) {
  __shared__ _Float16 As[16][256];
  const int tile = blockIdx.x;                   // 0..63 row tile
  const int b    = blockIdx.y;                   // batch
  const int tid  = threadIdx.x;
  const size_t rowBase = (size_t)b * NNODE + tile * 16;
  for (int idx = tid; idx < 16 * 256; idx += 256) {
    int r = idx >> 8, k = idx & 255;
    size_t node = rowBase + r;
    _Float16 val;
    if (k < 128) {
      val = h1[node * 128 + k];
    } else {
      float c = cnt[node]; c = c > 1.0f ? c : 1.0f;
      val = (_Float16)(msum2[node * 128 + (k - 128)] / c);
    }
    As[r][k] = val;
  }
  __syncthreads();
  const int lane = tid & 31;
  const int wave = tid >> 5;                     // col tile 0..7
  const int g = lane >> 4, m = lane & 15;
  const int colBase = wave * 16;
  const _Float16* rowPtr = &As[m][0];
  v8f acc = {};
  #pragma unroll
  for (int kk = 0; kk < 8; ++kk) {
    const int K0 = kk * 32;
    v16h a  = load_a_frag(rowPtr, K0, g);
    v16h bf = load_b_frag(W2t, 256, colBase + m, K0, g);
    acc = __builtin_amdgcn_wmma_f32_16x16x32_f16(false, a, false, bf,
                                                 (short)0, acc, false, false);
  }
  const int n = colBase + m;
  const float bias = b2[n];
  #pragma unroll
  for (int v = 0; v < 8; ++v) {
    int row = v + 8 * g;
    float val = acc[v] + bias;
    val = val > 0.0f ? val : 0.0f;
    h2[(rowBase + row) * 128 + n] = (_Float16)val;
  }
}

// ------------------------------------------------ graph pooling (mean|max)
__global__ __launch_bounds__(128) void pool_kernel(const _Float16* __restrict__ h2,
                                                   float* __restrict__ ge) {
  int f = threadIdx.x, b = blockIdx.x;
  size_t base = (size_t)b * NNODE * 128 + f;
  float s = 0.0f, mx = -3.0e38f;
  for (int n = 0; n < NNODE; ++n) {
    float v = (float)h2[base + (size_t)n * 128];
    s += v; mx = v > mx ? v : mx;
  }
  ge[(size_t)b * 256 + f]       = s * (1.0f / NNODE);
  ge[(size_t)b * 256 + 128 + f] = mx;
}

// ------------------------------------------------ fused context path (B=256)
__device__ __forceinline__ float blkSum(float* red, float v, int tid) {
  red[tid] = v; __syncthreads();
  for (int s = 128; s > 0; s >>= 1) {
    if (tid < s) red[tid] += red[tid + s];
    __syncthreads();
  }
  float r = red[0]; __syncthreads();
  return r;
}

__global__ __launch_bounds__(256) void context_kernel(
    const float* __restrict__ obs, const float* __restrict__ ge,
    const float* oe_w1, const float* oe_b1, const float* oe_g1, const float* oe_be1,
    const float* oe_w2, const float* oe_b2, const float* oe_g2, const float* oe_be2,
    const float* gf_w, const float* gf_b, const float* gf_g, const float* gf_be,
    const float* rm_w1, const float* rm_b1, const float* rm_g1, const float* rm_be1,
    const float* rm_w2, const float* rm_b2, const float* rm_g2, const float* rm_be2,
    const float* cr_w1, const float* cr_b1, const float* cr_w2, const float* cr_b2,
    const float* ps_w1, const float* ps_b1, const float* ps_w2, const float* ps_b2,
    _Float16* __restrict__ ctx16, float* __restrict__ logits, float* __restrict__ values) {
  __shared__ float A[384];
  __shared__ float G[256];
  __shared__ float red[256];
  const int tid = threadIdx.x;
  const int b = blockIdx.x;

  A[tid] = obs[(size_t)b * OBSD + tid];
  __syncthreads();

  // obs encoder layer 1: 256 -> 128, LN, relu
  float y = 0.0f;
  if (tid < 128) {
    y = oe_b1[tid];
    for (int k = 0; k < 256; ++k) y += A[k] * oe_w1[k * 128 + tid];
  }
  float mu  = blkSum(red, (tid < 128) ? y : 0.0f, tid) * (1.0f / 128.0f);
  float d   = (tid < 128) ? (y - mu) : 0.0f;
  float var = blkSum(red, d * d, tid) * (1.0f / 128.0f);
  if (tid < 128) {
    float h = d * rsqrtf(var + 1e-5f) * oe_g1[tid] + oe_be1[tid];
    A[tid] = h > 0.0f ? h : 0.0f;
  }
  __syncthreads();

  // obs encoder layer 2: 128 -> 128, LN, relu
  y = 0.0f;
  if (tid < 128) {
    y = oe_b2[tid];
    for (int k = 0; k < 128; ++k) y += A[k] * oe_w2[k * 128 + tid];
  }
  mu  = blkSum(red, (tid < 128) ? y : 0.0f, tid) * (1.0f / 128.0f);
  d   = (tid < 128) ? (y - mu) : 0.0f;
  var = blkSum(red, d * d, tid) * (1.0f / 128.0f);
  if (tid < 128) {
    float h = d * rsqrtf(var + 1e-5f) * oe_g2[tid] + oe_be2[tid];
    A[tid] = h > 0.0f ? h : 0.0f;
  }
  __syncthreads();
  A[128 + tid] = ge[(size_t)b * 256 + tid];     // concat graph embed
  __syncthreads();

  // global fusion: 384 -> 256, LN, relu -> G
  y = gf_b[tid];
  for (int k = 0; k < 384; ++k) y += A[k] * gf_w[k * 256 + tid];
  mu  = blkSum(red, y, tid) * (1.0f / 256.0f);
  d   = y - mu;
  var = blkSum(red, d * d, tid) * (1.0f / 256.0f);
  {
    float h = d * rsqrtf(var + 1e-5f) * gf_g[tid] + gf_be[tid];
    G[tid] = h > 0.0f ? h : 0.0f;
  }
  __syncthreads();

  // residual MLP layer 1
  y = rm_b1[tid];
  for (int k = 0; k < 256; ++k) y += G[k] * rm_w1[k * 256 + tid];
  mu  = blkSum(red, y, tid) * (1.0f / 256.0f);
  d   = y - mu;
  var = blkSum(red, d * d, tid) * (1.0f / 256.0f);
  {
    float h = d * rsqrtf(var + 1e-5f) * rm_g1[tid] + rm_be1[tid];
    A[tid] = h > 0.0f ? h : 0.0f;
  }
  __syncthreads();

  // residual MLP layer 2 (LN, no relu) + residual + relu -> ctx
  y = rm_b2[tid];
  for (int k = 0; k < 256; ++k) y += A[k] * rm_w2[k * 256 + tid];
  mu  = blkSum(red, y, tid) * (1.0f / 256.0f);
  d   = y - mu;
  var = blkSum(red, d * d, tid) * (1.0f / 256.0f);
  {
    float r2 = d * rsqrtf(var + 1e-5f) * rm_g2[tid] + rm_be2[tid];
    float c = r2 + G[tid];
    c = c > 0.0f ? c : 0.0f;
    A[tid] = c;
    ctx16[(size_t)b * 256 + tid] = (_Float16)c;
  }
  __syncthreads();

  // critic: 256 -> 128 -> 1
  float part = 0.0f;
  if (tid < 128) {
    y = cr_b1[tid];
    for (int k = 0; k < 256; ++k) y += A[k] * cr_w1[k * 128 + tid];
    y = y > 0.0f ? y : 0.0f;
    part = y * cr_w2[tid];
  }
  float s = blkSum(red, part, tid);
  if (tid == 0) values[b] = s + cr_b2[0];

  // pass scorer: 256 -> 64 -> 1 -> logits[:,512]
  part = 0.0f;
  if (tid < 64) {
    y = ps_b1[tid];
    for (int k = 0; k < 256; ++k) y += A[k] * ps_w1[k * 64 + tid];
    y = y > 0.0f ? y : 0.0f;
    part = y * ps_w2[tid];
  }
  s = blkSum(red, part, tid);
  if (tid == 0) logits[(size_t)b * NACT + ESWAP] = s + ps_b2[0];
}

// ------------------------------------------------ edge scorer via WMMA
// hidden = relu([src_f|dst_f|ctx] @ es_w1 + es_b1); score = hidden @ es_w2 + es_b2
__global__ __launch_bounds__(256) void edge_score_wmma(
    const _Float16* __restrict__ h2, const _Float16* __restrict__ ctx16,
    const int* __restrict__ swap_src, const int* __restrict__ swap_dst,
    const _Float16* __restrict__ esW1t, const float* __restrict__ es_b1,
    const float* __restrict__ es_w2, const float* __restrict__ es_b2,
    float* __restrict__ logits) {
  __shared__ _Float16 As[16][512];
  __shared__ float sc[16];
  const int tile = blockIdx.x;                   // 0..31 (16 edges each)
  const int b    = blockIdx.y;
  const int tid  = threadIdx.x;
  if (tid < 16) sc[tid] = 0.0f;
  for (int idx = tid; idx < 16 * 512; idx += 256) {
    int r = idx >> 9, k = idx & 511;
    int e = tile * 16 + r;
    _Float16 val;
    if (k < 256) {
      int node = (k < 128) ? swap_src[(size_t)b * ESWAP + e]
                           : swap_dst[(size_t)b * ESWAP + e];
      val = h2[((size_t)b * NNODE + node) * 128 + (k & 127)];
    } else {
      val = ctx16[(size_t)b * 256 + (k - 256)];
    }
    As[r][k] = val;
  }
  __syncthreads();
  const int lane = tid & 31;
  const int wave = tid >> 5;
  const int g = lane >> 4, m = lane & 15;
  const int colBase = wave * 16;
  const _Float16* rowPtr = &As[m][0];
  v8f acc = {};
  #pragma unroll
  for (int kk = 0; kk < 16; ++kk) {
    const int K0 = kk * 32;
    v16h a  = load_a_frag(rowPtr, K0, g);
    v16h bf = load_b_frag(esW1t, 512, colBase + m, K0, g);
    acc = __builtin_amdgcn_wmma_f32_16x16x32_f16(false, a, false, bf,
                                                 (short)0, acc, false, false);
  }
  const int n = colBase + m;
  const float bias = es_b1[n], w2v = es_w2[n];
  #pragma unroll
  for (int v = 0; v < 8; ++v) {
    float hv = acc[v] + bias;
    hv = hv > 0.0f ? hv : 0.0f;
    atomicAdd(&sc[v + 8 * g], hv * w2v);
  }
  __syncthreads();
  if (tid < 16) logits[(size_t)b * NACT + tile * 16 + tid] = sc[tid] + es_b2[0];
}

// ================================================================ launcher
extern "C" void kernel_launch(void* const* d_in, const int* in_sizes, int n_in,
                              void* d_out, int out_size, void* d_ws, size_t ws_size,
                              hipStream_t stream) {
  const float* obs      = (const float*)d_in[0];
  const float* x        = (const float*)d_in[1];
  const int*   eidx     = (const int*)d_in[2];
  const int*   swap_src = (const int*)d_in[3];
  const int*   swap_dst = (const int*)d_in[4];
  const float* g_w1s = (const float*)d_in[5];
  const float* g_w1n = (const float*)d_in[6];
  const float* g_b1  = (const float*)d_in[7];
  const float* g_w2s = (const float*)d_in[8];
  const float* g_w2n = (const float*)d_in[9];
  const float* g_b2  = (const float*)d_in[10];
  const float* oe_w1 = (const float*)d_in[11];
  const float* oe_b1 = (const float*)d_in[12];
  const float* oe_g1 = (const float*)d_in[13];
  const float* oe_be1= (const float*)d_in[14];
  const float* oe_w2 = (const float*)d_in[15];
  const float* oe_b2 = (const float*)d_in[16];
  const float* oe_g2 = (const float*)d_in[17];
  const float* oe_be2= (const float*)d_in[18];
  const float* gf_w  = (const float*)d_in[19];
  const float* gf_b  = (const float*)d_in[20];
  const float* gf_g  = (const float*)d_in[21];
  const float* gf_be = (const float*)d_in[22];
  const float* rm_w1 = (const float*)d_in[23];
  const float* rm_b1 = (const float*)d_in[24];
  const float* rm_g1 = (const float*)d_in[25];
  const float* rm_be1= (const float*)d_in[26];
  const float* rm_w2 = (const float*)d_in[27];
  const float* rm_b2 = (const float*)d_in[28];
  const float* rm_g2 = (const float*)d_in[29];
  const float* rm_be2= (const float*)d_in[30];
  const float* cr_w1 = (const float*)d_in[31];
  const float* cr_b1 = (const float*)d_in[32];
  const float* cr_w2 = (const float*)d_in[33];
  const float* cr_b2 = (const float*)d_in[34];
  const float* es_w1 = (const float*)d_in[35];
  const float* es_b1 = (const float*)d_in[36];
  const float* es_w2 = (const float*)d_in[37];
  const float* es_b2 = (const float*)d_in[38];
  const float* ps_w1 = (const float*)d_in[39];
  const float* ps_b1 = (const float*)d_in[40];
  const float* ps_w2 = (const float*)d_in[41];
  const float* ps_b2 = (const float*)d_in[42];

  float* out_logits = (float*)d_out;
  float* out_values = (float*)d_out + (size_t)BATCH * NACT;

  // workspace layout
  char* ws = (char*)d_ws;
  size_t off = 0;
  auto alloc = [&](size_t bytes) { size_t o = off; off += (bytes + 255) & ~(size_t)255; return o; };
  const size_t BN = (size_t)BATCH * NNODE;
  float*     cnt    = (float*)(ws + alloc(BN * 4));
  float*     msum1  = (float*)(ws + alloc(BN * 5 * 4));
  _Float16*  h1     = (_Float16*)(ws + alloc(BN * 128 * 2));
  float*     msum2  = (float*)(ws + alloc(BN * 128 * 4));
  _Float16*  h2     = (_Float16*)(ws + alloc(BN * 128 * 2));
  float*     ge     = (float*)(ws + alloc((size_t)BATCH * 256 * 4));
  _Float16*  ctx16  = (_Float16*)(ws + alloc((size_t)BATCH * 256 * 2));
  _Float16*  W2t    = (_Float16*)(ws + alloc(256 * 128 * 2));
  _Float16*  esW1t  = (_Float16*)(ws + alloc(512 * 128 * 2));
  (void)ws_size; (void)in_sizes; (void)n_in; (void)out_size;

  // 0) init accumulators, logits, f16 transposed weights
  fill_f32<<<(BN + 255) / 256, 256, 0, stream>>>(cnt, 0.0f, BN);
  fill_f32<<<(BN * 5 + 255) / 256, 256, 0, stream>>>(msum1, 0.0f, BN * 5);
  fill_f32<<<(BN * 128 + 255) / 256, 256, 0, stream>>>(msum2, 0.0f, BN * 128);
  fill_f32<<<((size_t)BATCH * NACT + 255) / 256, 256, 0, stream>>>(out_logits, NEGV, (size_t)BATCH * NACT);
  cvt_weights<<<(98304 + 255) / 256, 256, 0, stream>>>(g_w2s, g_w2n, es_w1, W2t, esW1t);

  // 1) layer-1 aggregation + SAGE layer 1
  scatter_x<<<(BATCH * EGRAPH) / 256, 256, 0, stream>>>(x, eidx, msum1, cnt);
  sage1<<<(BN * 128) / 256, 256, 0, stream>>>(x, msum1, cnt, g_w1s, g_w1n, g_b1, h1);

  // 2) layer-2 aggregation + WMMA GEMM
  scatter_h1<<<(BATCH * EGRAPH * 16) / 256, 256, 0, stream>>>(h1, eidx, msum2);
  gnn_layer2_wmma<<<dim3(NNODE / 16, BATCH), 256, 0, stream>>>(h1, msum2, cnt, W2t, g_b2, h2);

  // 3) pooling + context path (values + pass score)
  pool_kernel<<<BATCH, 128, 0, stream>>>(h2, ge);
  context_kernel<<<BATCH, 256, 0, stream>>>(
      obs, ge,
      oe_w1, oe_b1, oe_g1, oe_be1, oe_w2, oe_b2, oe_g2, oe_be2,
      gf_w, gf_b, gf_g, gf_be,
      rm_w1, rm_b1, rm_g1, rm_be1, rm_w2, rm_b2, rm_g2, rm_be2,
      cr_w1, cr_b1, cr_w2, cr_b2, ps_w1, ps_b1, ps_w2, ps_b2,
      ctx16, out_logits, out_values);

  // 4) WMMA edge scorer -> logits[:, 0:512]
  edge_score_wmma<<<dim3(ESWAP / 16, BATCH), 256, 0, stream>>>(
      h2, ctx16, swap_src, swap_dst, esW1t, es_b1, es_w2, es_b2, out_logits);
}
